// AttentionBlock_29111288332954
// MI455X (gfx1250) — matrix operand order repaired and checked by reading, not compiled, not run on hardware
//
#include <hip/hip_runtime.h>

// ---------------------------------------------------------------------------
// AttentionBlock for MI455X (gfx1250, wave32, WMMA bf16 16x16x32, fp32 accum)
//
//   1) gn_stats   : per-(batch,group) mean/rstd
//   2) gn_apply   : h fp32 -> d_out (residual base), h bf16 -> ws
//   3) wt_conv    : W^T bf16 for wq/wk/wv (B-matrix wants K contiguous)
//   4) qkv_gemm   : q,k bf16 [B*N,C]; v bf16 stored TRANSPOSED [B][C][N]
//                   (async-to-LDS double-buffered tiles, 1/sqrt(C) folded
//                    into q at store time)
//   5) attn       : flash attention Br=Bc=64; all tiles via
//                   global_load_async_to_lds_b128 with split s_wait_asynccnt
//                   so K(t+1)/V(t+1) loads overlap softmax / S-phase compute.
// ---------------------------------------------------------------------------

typedef unsigned short u16;
typedef __attribute__((ext_vector_type(16))) __bf16 v16bf;
typedef __attribute__((ext_vector_type(8)))  float  v8f;

union V16 { uint4 q[2]; v16bf v; };
union U8x16 { u16 s[8]; uint4 q; };

static __device__ __forceinline__ u16 f2bs(float f) {
    __bf16 h = (__bf16)f;
    return __builtin_bit_cast(u16, h);
}

static __device__ __forceinline__ v8f wmma_bf16(v16bf a, v16bf b, v8f c) {
    return __builtin_amdgcn_wmma_f32_16x16x32_bf16(
        false, a, false, b, (short)0, c, false, false);
}

// ---- CDNA5 async global->LDS copy (ASYNCcnt-tracked, no VGPR payload) -----
static __device__ __forceinline__ void async_copy16(void* lds, const void* g) {
    asm volatile("global_load_async_to_lds_b128 %0, %1, off"
                 :: "v"((unsigned)(unsigned long long)lds), "v"(g)
                 : "memory");
}
template <int N>
static __device__ __forceinline__ void wait_async() {
#if __has_builtin(__builtin_amdgcn_s_wait_asynccnt)
    __builtin_amdgcn_s_wait_asynccnt(N);
#else
    asm volatile("s_wait_asynccnt %0" :: "i"(N) : "memory");
#endif
}

constexpr int BATCH = 8;
constexpr int NPIX  = 4096;   // 64*64
constexpr int C     = 512;
constexpr int GRP   = 16;
constexpr int CG    = C / GRP;
constexpr float EPS = 1e-3f;

// ---------------------------------------------------------------------------
// 1) GroupNorm statistics
// ---------------------------------------------------------------------------
__global__ __launch_bounds__(256) void gn_stats(const float* __restrict__ x,
                                                float* __restrict__ stats) {
    const int bg = blockIdx.x;
    const int b = bg >> 4, g = bg & 15;
    const float* base = x + (size_t)b * NPIX * C + g * CG;

    float s = 0.f, s2 = 0.f;
    for (int t = threadIdx.x; t < NPIX * 4; t += 256) {
        const int p = t >> 2, ch = (t & 3) * 8;
        const float4* p4 = (const float4*)(base + (size_t)p * C + ch);
        float4 a = p4[0], c4 = p4[1];
        s  += a.x + a.y + a.z + a.w + c4.x + c4.y + c4.z + c4.w;
        s2 += a.x*a.x + a.y*a.y + a.z*a.z + a.w*a.w
            + c4.x*c4.x + c4.y*c4.y + c4.z*c4.z + c4.w*c4.w;
    }
    for (int o = 16; o; o >>= 1) {
        s  += __shfl_down(s,  o, 32);
        s2 += __shfl_down(s2, o, 32);
    }
    __shared__ float rs[8], rs2[8];
    if ((threadIdx.x & 31) == 0) { rs[threadIdx.x >> 5] = s; rs2[threadIdx.x >> 5] = s2; }
    __syncthreads();
    if (threadIdx.x == 0) {
        float S = 0.f, S2 = 0.f;
        for (int i = 0; i < 8; ++i) { S += rs[i]; S2 += rs2[i]; }
        const float inv = 1.f / (float)(NPIX * CG);
        const float mean = S * inv;
        const float var  = S2 * inv - mean * mean;
        stats[bg * 2 + 0] = mean;
        stats[bg * 2 + 1] = rsqrtf(var + EPS);
    }
}

// ---------------------------------------------------------------------------
// 2) Apply GroupNorm
// ---------------------------------------------------------------------------
__global__ __launch_bounds__(256) void gn_apply(const float* __restrict__ x,
                                                const float* __restrict__ gamma,
                                                const float* __restrict__ beta,
                                                const float* __restrict__ stats,
                                                float* __restrict__ out,
                                                u16* __restrict__ hbf) {
    const size_t idx = (size_t)blockIdx.x * 256 + threadIdx.x;
    const size_t e = idx * 4;
    const int c = (int)(e & (C - 1));
    const size_t n = e >> 9;
    const int b = (int)(n >> 12);
    const int g = c >> 5;
    const float mean = stats[(b * GRP + g) * 2 + 0];
    const float rstd = stats[(b * GRP + g) * 2 + 1];

    float4 xv = *(const float4*)(x + e);
    float4 gv = *(const float4*)(gamma + c);
    float4 bv = *(const float4*)(beta + c);
    float4 h;
    h.x = (xv.x - mean) * rstd * gv.x + bv.x;
    h.y = (xv.y - mean) * rstd * gv.y + bv.y;
    h.z = (xv.z - mean) * rstd * gv.z + bv.z;
    h.w = (xv.w - mean) * rstd * gv.w + bv.w;
    *(float4*)(out + e) = h;
    ushort4 hb;
    hb.x = f2bs(h.x); hb.y = f2bs(h.y); hb.z = f2bs(h.z); hb.w = f2bs(h.w);
    *(ushort4*)(hbf + e) = hb;
}

// ---------------------------------------------------------------------------
// 3) Transpose+convert weight: wt[o*512 + i] = bf16(W[i*512 + o])
// ---------------------------------------------------------------------------
__global__ __launch_bounds__(256) void wt_conv(const float* __restrict__ w,
                                               u16* __restrict__ wt) {
    const int idx = blockIdx.x * 256 + threadIdx.x;
    const int o = idx >> 9, i = idx & (C - 1);
    wt[(size_t)o * C + i] = f2bs(w[(size_t)i * C + o]);
}

// ---------------------------------------------------------------------------
// 4) QKV GEMM, async double-buffered. transpose_v=1 stores dst as [B][C][N].
// ---------------------------------------------------------------------------
constexpr int GP = 40;                       // padded K-stride for 32-wide tiles
constexpr int TB = 5120;                     // elems per tile buffer (128*GP)
constexpr size_t GEMM_SMEM = 4 * TB * 2;     // 40960 B (sO overlays this)

__global__ __launch_bounds__(256) void qkv_gemm(const u16* __restrict__ hbf,
                                                const u16* __restrict__ wt,
                                                const float* __restrict__ bias,
                                                u16* __restrict__ dst,
                                                float scale, int transpose_v) {
    extern __shared__ __align__(16) char smem[];
    u16* sm = (u16*)smem;   // [0,TB)=A0 [TB,2TB)=A1 [2TB,3TB)=B0 [3TB,4TB)=B1

    const int tid  = threadIdx.x;
    const int wave = tid >> 5, lane = tid & 31;
    const int hi   = lane >> 4, lr = lane & 15;
    const int m0 = blockIdx.x * 128;
    const int n0 = blockIdx.y * 128;

    auto issue = [&](int kb, int bi) {       // fully unrolled: no exec-mask loop
        u16* A  = sm + bi * TB;
        u16* Bp = sm + (2 + bi) * TB;
        const int r  = tid >> 2, ch = tid & 3;
        const int r2 = r + 64;
        async_copy16(&A [r  * GP + ch * 8], &hbf[((size_t)(m0 + r )) * C + kb * 32 + ch * 8]);
        async_copy16(&Bp[r  * GP + ch * 8], &wt [((size_t)(n0 + r )) * C + kb * 32 + ch * 8]);
        async_copy16(&A [r2 * GP + ch * 8], &hbf[((size_t)(m0 + r2)) * C + kb * 32 + ch * 8]);
        async_copy16(&Bp[r2 * GP + ch * 8], &wt [((size_t)(n0 + r2)) * C + kb * 32 + ch * 8]);
    };

    v8f acc[8];
    v8f z8; for (int i = 0; i < 8; ++i) z8[i] = 0.f;
    for (int t = 0; t < 8; ++t) acc[t] = z8;

    issue(0, 0);
    wait_async<0>();
    __syncthreads();

    for (int kb = 0; kb < C / 32; ++kb) {
        const int cur = kb & 1;
        issue((kb + 1) & 15, cur ^ 1);       // prefetch overlaps compute

        const u16* A  = sm + cur * TB;
        const u16* Bp = sm + (2 + cur) * TB;
        V16 a;
        const u16* pa = &A[(wave * 16 + lr) * GP + hi * 8];
        a.q[0] = *(const uint4*)pa;
        a.q[1] = *(const uint4*)(pa + 16);
#pragma unroll
        for (int tc = 0; tc < 8; ++tc) {
            V16 bm;
            const u16* pb = &Bp[(tc * 16 + lr) * GP + hi * 8];
            bm.q[0] = *(const uint4*)pb;
            bm.q[1] = *(const uint4*)(pb + 16);
            acc[tc] = wmma_bf16(a.v, bm.v, acc[tc]);
        }
        wait_async<0>();
        __syncthreads();
    }

    if (!transpose_v) {
#pragma unroll
        for (int tc = 0; tc < 8; ++tc)
#pragma unroll
            for (int r = 0; r < 8; ++r) {
                const int row = m0 + wave * 16 + r + 8 * hi;
                const int col = n0 + tc * 16 + lr;
                const float val = (acc[tc][r] + bias[col]) * scale;
                dst[(size_t)row * C + col] = f2bs(val);
            }
    } else {
        // stage tile in LDS, then coalesced column-major (transposed) stores
        u16* sO = sm;                        // 128 x 136, overlays tile buffers
#pragma unroll
        for (int tc = 0; tc < 8; ++tc)
#pragma unroll
            for (int r = 0; r < 8; ++r) {
                const int rowl = wave * 16 + r + 8 * hi;
                const int coll = tc * 16 + lr;
                sO[rowl * 136 + coll] = f2bs((acc[tc][r] + bias[n0 + coll]) * scale);
            }
        __syncthreads();
        const int b  = m0 >> 12;
        const int nb = m0 & (NPIX - 1);
#pragma unroll
        for (int it = 0; it < 8; ++it) {
            const int t = tid + it * 256;
            const int coll = t >> 4, chunk = t & 15;
            U8x16 pk;
#pragma unroll
            for (int i = 0; i < 8; ++i) pk.s[i] = sO[(chunk * 8 + i) * 136 + coll];
            *(uint4*)&dst[((size_t)b * C + n0 + coll) * NPIX + nb + chunk * 8] = pk.q;
        }
    }
}

// ---------------------------------------------------------------------------
// 5) Flash attention, async-pipelined tiles. v is [B][C][N] (pre-transposed).
// ---------------------------------------------------------------------------
constexpr int BR = 64;
constexpr int CP = 520;
constexpr int KP = 72;
constexpr int SP = 68;
constexpr int PP = 72;
constexpr size_t ATT_SMEM =
    (size_t)(2 * BR * CP + C * KP + BR * PP) * 2 + (size_t)BR * SP * 4 + 3 * BR * 4;

__global__ __launch_bounds__(256) void attn(const u16* __restrict__ q,
                                            const u16* __restrict__ k,
                                            const u16* __restrict__ vt,
                                            float* __restrict__ out) {
    extern __shared__ __align__(16) char smem[];
    u16*   sQ  = (u16*)smem;                 // [64][CP]
    u16*   sK  = sQ + BR * CP;               // [64][CP]
    u16*   sVt = sK + BR * CP;               // [C][KP]
    float* sS  = (float*)(sVt + C * KP);     // [64][SP]
    u16*   sP  = (u16*)(sS + BR * SP);       // [64][PP]
    float* sM  = (float*)(sP + BR * PP);
    float* sL  = sM + BR;
    float* sCr = sL + BR;

    const int tid  = threadIdx.x;
    const int wave = tid >> 5, lane = tid & 31;
    const int hi   = lane >> 4, lr = lane & 15;
    const int b  = blockIdx.x >> 6;
    const int m0 = (blockIdx.x & 63) * BR;
    const size_t baseQ = ((size_t)b * NPIX + m0) * C;
    const size_t baseK = (size_t)b * NPIX * C;
    const size_t baseV = (size_t)b * C * NPIX;
    const int d0 = wave * 64;

    const int rQ = tid >> 6, chQ = tid & 63;     // row/chunk split for Q/K tiles
    const int cV = tid >> 3, chV = tid & 7;      // row/chunk split for Vt tile

    auto issueK = [&](int kt) {   // 16 async instr / wave, constant strides
        u16* l = &sK[rQ * CP + chQ * 8];
        const u16* g = &k[baseK + ((size_t)kt * BR + rQ) * C + chQ * 8];
#pragma unroll
        for (int it = 0; it < 16; ++it)
            async_copy16(l + it * (4 * CP), g + it * (4 * C));
    };
    auto issueV = [&](int kt) {   // 16 async instr / wave
        u16* l = &sVt[cV * KP + chV * 8];
        const u16* g = &vt[baseV + (size_t)cV * NPIX + kt * BR + chV * 8];
#pragma unroll
        for (int it = 0; it < 16; ++it)
            async_copy16(l + it * (32 * KP), g + it * (size_t)(32 * NPIX));
    };

    // prologue: Q + K(0), wait; V(0) left in flight
    {
        u16* l = &sQ[rQ * CP + chQ * 8];
        const u16* g = &q[baseQ + (size_t)rQ * C + chQ * 8];
#pragma unroll
        for (int it = 0; it < 16; ++it)
            async_copy16(l + it * (4 * CP), g + it * (4 * C));
    }
    issueK(0);
    wait_async<0>();
    issueV(0);

    if (tid < BR) { sM[tid] = -1e30f; sL[tid] = 0.f; }
    v8f accO[4][4];
    v8f z8; for (int i = 0; i < 8; ++i) z8[i] = 0.f;
    for (int tr = 0; tr < 4; ++tr)
        for (int tc = 0; tc < 4; ++tc) accO[tr][tc] = z8;
    __syncthreads();

    for (int kt = 0; kt < NPIX / BR; ++kt) {
        const int ktn = (kt + 1) & 63;       // wrap: tail prefetch stays in bounds

        // ---- S = Q * K^T (scale folded into q) ----
        {
            const int str = wave >> 1;
            const int tc0 = (wave & 1) * 2;
            v8f accS[2] = { z8, z8 };
#pragma unroll 4
            for (int kc = 0; kc < C / 32; ++kc) {
                V16 a;
                const u16* pa = &sQ[(str * 16 + lr) * CP + kc * 32 + hi * 8];
                a.q[0] = *(const uint4*)pa;
                a.q[1] = *(const uint4*)(pa + 16);
#pragma unroll
                for (int t2 = 0; t2 < 2; ++t2) {
                    V16 bm;
                    const u16* pb = &sK[((tc0 + t2) * 16 + lr) * CP + kc * 32 + hi * 8];
                    bm.q[0] = *(const uint4*)pb;
                    bm.q[1] = *(const uint4*)(pb + 16);
                    accS[t2] = wmma_bf16(a.v, bm.v, accS[t2]);
                }
            }
#pragma unroll
            for (int t2 = 0; t2 < 2; ++t2)
#pragma unroll
                for (int r = 0; r < 8; ++r)
                    sS[(str * 16 + r + 8 * hi) * SP + (tc0 + t2) * 16 + lr] = accS[t2][r];
        }
        __syncthreads();                     // sS published; sK free

        issueK(ktn);                         // overlaps softmax + PV below

        // ---- online softmax ----
        {
            const int row = tid >> 2, seg = tid & 3;
            float* srow = &sS[row * SP + seg * 16];
            float tmax = -1e30f;
#pragma unroll
            for (int i = 0; i < 16; ++i) tmax = fmaxf(tmax, srow[i]);
            tmax = fmaxf(tmax, __shfl_xor(tmax, 1, 32));
            tmax = fmaxf(tmax, __shfl_xor(tmax, 2, 32));
            const float mo = sM[row];
            const float mn = fmaxf(mo, tmax);
            float ps = 0.f;
            u16* prow = &sP[row * PP + seg * 16];
#pragma unroll
            for (int i = 0; i < 16; ++i) {
                const float p = __expf(srow[i] - mn);
                ps += p;
                prow[i] = f2bs(p);
            }
            ps += __shfl_xor(ps, 1, 32);
            ps += __shfl_xor(ps, 2, 32);
            if (seg == 0) {
                const float corr = __expf(mo - mn);
                sL[row] = sL[row] * corr + ps;
                sM[row] = mn;
                sCr[row] = corr;
            }
        }
        wait_async<16>();                    // V(kt) done (issued before K(ktn))
        __syncthreads();                     // publish sVt + sP

        // ---- O = O*corr + P @ Vtile ----
        {
            float corr[4][8];
#pragma unroll
            for (int tr = 0; tr < 4; ++tr)
#pragma unroll
                for (int r = 0; r < 8; ++r)
                    corr[tr][r] = sCr[tr * 16 + r + 8 * hi];
#pragma unroll
            for (int tr = 0; tr < 4; ++tr)
#pragma unroll
                for (int tc = 0; tc < 4; ++tc)
#pragma unroll
                    for (int r = 0; r < 8; ++r)
                        accO[tr][tc][r] *= corr[tr][r];

#pragma unroll
            for (int kc2 = 0; kc2 < 2; ++kc2) {
                V16 A[4];
#pragma unroll
                for (int tr = 0; tr < 4; ++tr) {
                    const u16* pa = &sP[(tr * 16 + lr) * PP + kc2 * 32 + hi * 8];
                    A[tr].q[0] = *(const uint4*)pa;
                    A[tr].q[1] = *(const uint4*)(pa + 16);
                }
#pragma unroll
                for (int tc = 0; tc < 4; ++tc) {
                    V16 bm;
                    const u16* pb = &sVt[(d0 + tc * 16 + lr) * KP + kc2 * 32 + hi * 8];
                    bm.q[0] = *(const uint4*)pb;
                    bm.q[1] = *(const uint4*)(pb + 16);
#pragma unroll
                    for (int tr = 0; tr < 4; ++tr)
                        accO[tr][tc] = wmma_bf16(A[tr].v, bm.v, accO[tr][tc]);
                }
            }
        }
        __syncthreads();                     // sVt free

        issueV(ktn);                         // overlaps next S-phase
        wait_async<16>();                    // K(ktn) done (issued first)
        __syncthreads();                     // publish sK
    }

    // ---- epilogue: O /= l, fused residual add ----
    float li[4][8];
#pragma unroll
    for (int tr = 0; tr < 4; ++tr)
#pragma unroll
        for (int r = 0; r < 8; ++r)
            li[tr][r] = 1.f / sL[tr * 16 + r + 8 * hi];

#pragma unroll
    for (int tr = 0; tr < 4; ++tr)
#pragma unroll
        for (int tc = 0; tc < 4; ++tc)
#pragma unroll
            for (int r = 0; r < 8; ++r) {
                const int row = m0 + tr * 16 + r + 8 * hi;
                const int col = d0 + tc * 16 + lr;
                const size_t oi = ((size_t)b * NPIX + row) * C + col;
                out[oi] += accO[tr][tc][r] * li[tr][r];
            }
}

// ---------------------------------------------------------------------------
// Host launcher
// ---------------------------------------------------------------------------
extern "C" void kernel_launch(void* const* d_in, const int* in_sizes, int n_in,
                              void* d_out, int out_size, void* d_ws, size_t ws_size,
                              hipStream_t stream) {
    (void)in_sizes; (void)n_in; (void)out_size; (void)ws_size;
    const float* x     = (const float*)d_in[0];
    const float* gamma = (const float*)d_in[1];
    const float* beta  = (const float*)d_in[2];
    const float* wq    = (const float*)d_in[3];
    const float* bq    = (const float*)d_in[4];
    const float* wk    = (const float*)d_in[5];
    const float* bk    = (const float*)d_in[6];
    const float* wv    = (const float*)d_in[7];
    const float* bv    = (const float*)d_in[8];
    float* out = (float*)d_out;

    char* w = (char*)d_ws;
    float* stats = (float*)w;
    size_t off = 1024;
    u16* wtq = (u16*)(w + off); off += (size_t)C * C * 2;
    u16* wtk = (u16*)(w + off); off += (size_t)C * C * 2;
    u16* wtv = (u16*)(w + off); off += (size_t)C * C * 2;
    u16* hbf = (u16*)(w + off); off += (size_t)BATCH * NPIX * C * 2;
    u16* qb  = (u16*)(w + off); off += (size_t)BATCH * NPIX * C * 2;
    u16* kbuf= (u16*)(w + off); off += (size_t)BATCH * NPIX * C * 2;
    u16* vtb = (u16*)(w + off); off += (size_t)BATCH * NPIX * C * 2;  // [B][C][N]

    gn_stats<<<BATCH * GRP, 256, 0, stream>>>(x, stats);
    gn_apply<<<(BATCH * NPIX * C / 4) / 256, 256, 0, stream>>>(x, gamma, beta, stats, out, hbf);
    wt_conv<<<(C * C) / 256, 256, 0, stream>>>(wq, wtq);
    wt_conv<<<(C * C) / 256, 256, 0, stream>>>(wk, wtk);
    wt_conv<<<(C * C) / 256, 256, 0, stream>>>(wv, wtv);

    const float scale = 0.044194173824159216f;  // 512^-0.5, folded into q
    dim3 ggrid(BATCH * NPIX / 128, C / 128);
    qkv_gemm<<<ggrid, 256, GEMM_SMEM, stream>>>(hbf, wtq, bq, qb,  scale, 0);
    qkv_gemm<<<ggrid, 256, GEMM_SMEM, stream>>>(hbf, wtk, bk, kbuf, 1.f,  0);
    qkv_gemm<<<ggrid, 256, GEMM_SMEM, stream>>>(hbf, wtv, bv, vtb,  1.f,  1);

    attn<<<BATCH * (NPIX / BR), 256, ATT_SMEM, stream>>>(qb, kbuf, vtb, out);
}
